// Distiller_31782757990894
// MI455X (gfx1250) — matrix operand order repaired and checked by reading, not compiled, not run on hardware
//
#include <hip/hip_runtime.h>
#include <hip/hip_bf16.h>

typedef __attribute__((ext_vector_type(16))) __bf16 v16bf;
typedef __attribute__((ext_vector_type(8)))  float  v8f;

#define BQ  16384   // batch
#define DD  768     // input dim
#define HH  1024    // expert hidden
#define EE  6       // experts
#define GHD 1024    // gating hidden

// GEMM tiling
#define BM 128
#define BN 128
#define BK 32
#define LDK (BK + 8)   // LDS stride (elements) for K-major tiles: 80B rows, 16B aligned

// ---------------------------------------------------------------------------
// Async global->LDS copy (CDNA5 GLOBAL_LOAD_ASYNC_TO_LDS_B128, ASYNCcnt)
// GVS addressing: saddr(SGPR pair) + vaddr(i32) ; dest = LDS-offset VGPR.
// ---------------------------------------------------------------------------
__device__ __forceinline__ void async_copy_b128(void* lds_dst, const void* gbase,
                                                unsigned goff_bytes) {
    unsigned lds_off = (unsigned)(uintptr_t)lds_dst;  // LDS aperture: low 32 bits = offset
    asm volatile("global_load_async_to_lds_b128 %0, %1, %2"
                 :
                 : "v"(lds_off), "v"(goff_bytes),
                   "s"((unsigned long long)(uintptr_t)gbase)
                 : "memory");
}
__device__ __forceinline__ void wait_async0() {
    asm volatile("s_wait_asynccnt 0" ::: "memory");
}

// ---------------------------------------------------------------------------
// f32 -> bf16 conversion (row-major, for A operands)
// ---------------------------------------------------------------------------
__global__ __launch_bounds__(256) void f32_to_bf16_kernel(const float* __restrict__ in,
                                                          __bf16* __restrict__ out, int n) {
    int i = blockIdx.x * 256 + threadIdx.x;
    if (i < n) out[i] = (__bf16)in[i];
}

// ---------------------------------------------------------------------------
// f32 [K,N] -> bf16 [N,K] tiled transpose-convert (weights become K-major so
// WMMA B fragments are contiguous b128 LDS loads). blockIdx.z = expert.
// ---------------------------------------------------------------------------
__global__ __launch_bounds__(256)
void transpose_f32_to_bf16(const float* __restrict__ in, __bf16* __restrict__ out,
                           int K, int N) {
    __shared__ float tile[32][33];
    const size_t eoff = (size_t)blockIdx.z * K * N;
    const float* ine = in + eoff;
    __bf16* oute = out + eoff;
    int n0 = blockIdx.x * 32, k0 = blockIdx.y * 32;
    int tx = threadIdx.x & 31, ty = threadIdx.x >> 5;  // 32 x 8
#pragma unroll
    for (int i = 0; i < 32; i += 8)
        tile[ty + i][tx] = ine[(size_t)(k0 + ty + i) * N + n0 + tx];
    __syncthreads();
#pragma unroll
    for (int i = 0; i < 32; i += 8)
        oute[(size_t)(n0 + ty + i) * K + k0 + tx] = (__bf16)tile[tx][ty + i];
}

// ---------------------------------------------------------------------------
// Tiled bf16 WMMA GEMM: C[e] = act(A[e] (MxK, row-major) * Bt[e] (NxK, K-major) + bias[e])
// 256 threads = 8 waves; wave grid 2(M)x4(N); each wave 64x32 via 4x2 16x16 tiles.
// Double-buffered LDS fed by async global->LDS b128 copies; 1 barrier / K-step.
// ---------------------------------------------------------------------------
template<bool RELU, bool OUT_BF16>
__global__ __launch_bounds__(256)
void gemm_bf16_wmma(const __bf16* __restrict__ A,  long saA,
                    const __bf16* __restrict__ Bt, long saB,
                    const float*  __restrict__ bias, long saBias,
                    void* __restrict__ Cout, long saC,
                    int lda, int ldc, int K)
{
    __shared__ __bf16 ldsA[2][BM * LDK];
    __shared__ __bf16 ldsB[2][BN * LDK];

    const int e = blockIdx.z;
    const __bf16* Ab = A  + (size_t)e * (size_t)saA;
    const __bf16* Bb = Bt + (size_t)e * (size_t)saB;
    const float*  bb = bias + (size_t)e * (size_t)saBias;

    const int t     = threadIdx.x;
    const int lane  = t & 31;
    const int w     = t >> 5;
    const int waveM = w >> 2;      // 0..1
    const int waveN = w & 3;       // 0..3
    const int half  = lane >> 4;   // 0..1
    const int r     = lane & 15;   // 0..15

    const int rowBase = blockIdx.y * BM;
    const int colBase = blockIdx.x * BN;

    v8f acc[4][2];
#pragma unroll
    for (int mt = 0; mt < 4; ++mt)
#pragma unroll
        for (int nt = 0; nt < 2; ++nt)
#pragma unroll
            for (int i = 0; i < 8; ++i) acc[mt][nt][i] = 0.0f;

    // per-thread static slice of the tile copies (4 x b128 per thread per step)
    const int rowA = t >> 2;            // 0..63 (+64 on second pass)
    const int colA = (t & 3) << 3;      // 0,8,16,24

    auto issueTile = [&](int buf, int k0) {
#pragma unroll
        for (int p = 0; p < 2; ++p) {
            int row = rowA + p * 64;
            async_copy_b128(&ldsA[buf][row * LDK + colA], Ab,
                            (unsigned)(((rowBase + row) * lda + k0 + colA) * 2));
        }
#pragma unroll
        for (int p = 0; p < 2; ++p) {
            int n = rowA + p * 64;
            async_copy_b128(&ldsB[buf][n * LDK + colA], Bb,
                            (unsigned)(((colBase + n) * K + k0 + colA) * 2));
        }
    };

    const int nk = K / BK;
    issueTile(0, 0);

    for (int kt = 0; kt < nk; ++kt) {
        wait_async0();          // own async copies for tile kt have landed
        __syncthreads();        // everyone's copies landed; prior compute done
        if (kt + 1 < nk) issueTile((kt + 1) & 1, (kt + 1) * BK);

        const __bf16* bufA = ldsA[kt & 1];
        const __bf16* bufB = ldsB[kt & 1];

        union U16 { uint4 q[2]; v16bf v; };

        // A fragments (16-bit A 16x32 layout, ISA 7.12.2): two b128 per tile
        v16bf afrag[4];
#pragma unroll
        for (int mt = 0; mt < 4; ++mt) {
            const __bf16* ar = bufA + (waveM * 64 + mt * 16 + r) * LDK;
            U16 u;
            u.q[0] = *reinterpret_cast<const uint4*>(ar + 8 * half);       // K = 8h..8h+7
            u.q[1] = *reinterpret_cast<const uint4*>(ar + 16 + 8 * half);  // K = 16+8h..
            afrag[mt] = u.v;
        }
        // B fragments: lane holds column (n = r), K = 8h..8h+7, 16+8h..16+8h+7
        v16bf bfrag[2];
#pragma unroll
        for (int nt = 0; nt < 2; ++nt) {
            const __bf16* br = bufB + (waveN * 32 + nt * 16 + r) * LDK;
            U16 u;
            u.q[0] = *reinterpret_cast<const uint4*>(br + 8 * half);
            u.q[1] = *reinterpret_cast<const uint4*>(br + 16 + 8 * half);
            bfrag[nt] = u.v;
        }

#pragma unroll
        for (int mt = 0; mt < 4; ++mt)
#pragma unroll
            for (int nt = 0; nt < 2; ++nt)
                acc[mt][nt] = __builtin_amdgcn_wmma_f32_16x16x32_bf16(
                    false, afrag[mt], false, bfrag[nt],
                    (short)0, acc[mt][nt], false, false);
    }

    // epilogue: C layout VGPR i -> row (i + 8*half), col r
#pragma unroll
    for (int mt = 0; mt < 4; ++mt) {
#pragma unroll
        for (int nt = 0; nt < 2; ++nt) {
            int colg = colBase + waveN * 32 + nt * 16 + r;
            float bi = bb[colg];
#pragma unroll
            for (int i = 0; i < 8; ++i) {
                int rowg = rowBase + waveM * 64 + mt * 16 + i + 8 * half;
                float v = acc[mt][nt][i] + bi;
                if (RELU) v = v > 0.0f ? v : 0.0f;
                size_t off = (size_t)e * (size_t)saC + (size_t)rowg * ldc + colg;
                if (OUT_BF16) ((__bf16*)Cout)[off] = (__bf16)v;
                else          ((float*)Cout)[off]  = v;
            }
        }
    }
}

// ---------------------------------------------------------------------------
// Gating tail: logits = gh @ Wg2 + bg2, softmax(E), plus half-softmaxes.
// ---------------------------------------------------------------------------
__global__ __launch_bounds__(256)
void gating_kernel(const __bf16* __restrict__ gh, const float* __restrict__ Wg2,
                   const float* __restrict__ bg2,
                   float* __restrict__ gating, float* __restrict__ w1n,
                   float* __restrict__ w2n)
{
    __shared__ float sW[GHD * EE];
    for (int i = threadIdx.x; i < GHD * EE; i += 256) sW[i] = Wg2[i];
    __syncthreads();

    int b = blockIdx.x * 256 + threadIdx.x;
    const __bf16* row = gh + (size_t)b * GHD;

    float acc[EE] = {0.f, 0.f, 0.f, 0.f, 0.f, 0.f};
    for (int k = 0; k < GHD; ++k) {
        float g = (float)row[k];
#pragma unroll
        for (int e2 = 0; e2 < EE; ++e2) acc[e2] += g * sW[k * EE + e2];
    }

    float m = -1e30f;
#pragma unroll
    for (int e2 = 0; e2 < EE; ++e2) { acc[e2] += bg2[e2]; m = fmaxf(m, acc[e2]); }
    float p[EE], s = 0.f;
#pragma unroll
    for (int e2 = 0; e2 < EE; ++e2) { p[e2] = __expf(acc[e2] - m); s += p[e2]; }
    float inv = 1.0f / s;
#pragma unroll
    for (int e2 = 0; e2 < EE; ++e2) {
        p[e2] *= inv;
        gating[(size_t)b * EE + e2] = p[e2];
    }

    float m1 = fmaxf(fmaxf(p[0], p[1]), p[2]);
    float m2 = fmaxf(fmaxf(p[3], p[4]), p[5]);
    float q1[3], q2[3], s1 = 0.f, s2 = 0.f;
#pragma unroll
    for (int e2 = 0; e2 < 3; ++e2) { q1[e2] = __expf(p[e2] - m1); s1 += q1[e2]; }
#pragma unroll
    for (int e2 = 0; e2 < 3; ++e2) { q2[e2] = __expf(p[3 + e2] - m2); s2 += q2[e2]; }
    float i1 = 1.0f / s1, i2 = 1.0f / s2;
#pragma unroll
    for (int e2 = 0; e2 < 3; ++e2) {
        w1n[(size_t)b * 3 + e2] = q1[e2] * i1;
        w2n[(size_t)b * 3 + e2] = q2[e2] * i2;
    }
}

// ---------------------------------------------------------------------------
// Combine: final / teacher1 / teacher2 gated sums over experts. float4 per thread.
// ---------------------------------------------------------------------------
__global__ __launch_bounds__(256)
void combine_kernel(const float* __restrict__ expert_out, const float* __restrict__ gating,
                    const float* __restrict__ w1n, const float* __restrict__ w2n,
                    float* __restrict__ fin, float* __restrict__ t1, float* __restrict__ t2)
{
    int idx = blockIdx.x * 256 + threadIdx.x;   // over B * (D/4)
    int b   = idx / (DD / 4);
    int d   = (idx % (DD / 4)) * 4;

    float f0 = 0, f1 = 0, f2 = 0, f3 = 0;
    float a0 = 0, a1 = 0, a2 = 0, a3 = 0;
    float c0 = 0, c1 = 0, c2 = 0, c3 = 0;
#pragma unroll
    for (int e = 0; e < EE; ++e) {
        const float4 v = *reinterpret_cast<const float4*>(
            &expert_out[((size_t)b * EE + e) * DD + d]);
        float ge = gating[(size_t)b * EE + e];
        f0 += v.x * ge; f1 += v.y * ge; f2 += v.z * ge; f3 += v.w * ge;
        if (e < 3) {
            float we = w1n[(size_t)b * 3 + e];
            a0 += v.x * we; a1 += v.y * we; a2 += v.z * we; a3 += v.w * we;
        } else {
            float we = w2n[(size_t)b * 3 + (e - 3)];
            c0 += v.x * we; c1 += v.y * we; c2 += v.z * we; c3 += v.w * we;
        }
    }
    *reinterpret_cast<float4*>(&fin[(size_t)b * DD + d]) = make_float4(f0, f1, f2, f3);
    *reinterpret_cast<float4*>(&t1 [(size_t)b * DD + d]) = make_float4(a0, a1, a2, a3);
    *reinterpret_cast<float4*>(&t2 [(size_t)b * DD + d]) = make_float4(c0, c1, c2, c3);
}

// ---------------------------------------------------------------------------
// Launch
// ---------------------------------------------------------------------------
extern "C" void kernel_launch(void* const* d_in, const int* in_sizes, int n_in,
                              void* d_out, int out_size, void* d_ws, size_t ws_size,
                              hipStream_t stream) {
    const float* x   = (const float*)d_in[0];
    const float* W1  = (const float*)d_in[1];
    const float* b1  = (const float*)d_in[2];
    const float* W2  = (const float*)d_in[3];
    const float* b2  = (const float*)d_in[4];
    const float* Wg1 = (const float*)d_in[5];
    const float* bg1 = (const float*)d_in[6];
    const float* Wg2 = (const float*)d_in[7];
    const float* bg2 = (const float*)d_in[8];

    float* out        = (float*)d_out;
    float* expert_out = out;                                  // [B,E,D]
    float* gating     = expert_out + (size_t)BQ * EE * DD;    // [B,E]
    float* fin        = gating + (size_t)BQ * EE;             // [B,D]
    float* t1         = fin + (size_t)BQ * DD;                // [B,D]
    float* t2         = t1 + (size_t)BQ * DD;                 // [B,D]

    size_t off = 0;
    auto take = [&](size_t bytes) {
        void* p = (char*)d_ws + off;
        off += (bytes + 255) & ~(size_t)255;
        return p;
    };
    __bf16* xbf   = (__bf16*)take((size_t)BQ * DD * 2);       // [B,D]
    __bf16* w1t   = (__bf16*)take((size_t)EE * HH * DD * 2);  // [e][H,D] (N,K)
    __bf16* w2t   = (__bf16*)take((size_t)EE * DD * HH * 2);  // [e][D,H] (N,K)
    __bf16* wg1t  = (__bf16*)take((size_t)GHD * DD * 2);      // [GH,D]  (N,K)
    __bf16* ghbf  = (__bf16*)take((size_t)BQ * GHD * 2);      // [B,GH]
    __bf16* hbf   = (__bf16*)take((size_t)EE * BQ * HH * 2);  // [e][B,H]
    float*  w1n   = (float*)take((size_t)BQ * 3 * 4);
    float*  w2n   = (float*)take((size_t)BQ * 3 * 4);
    (void)ws_size; (void)in_sizes; (void)n_in; (void)out_size;

    dim3 blk(256);

    // 1) precision conversion; weights transposed to (N,K) for b128 B-fragments
    f32_to_bf16_kernel<<<(BQ * DD + 255) / 256, blk, 0, stream>>>(x, xbf, BQ * DD);
    transpose_f32_to_bf16<<<dim3(HH / 32, DD / 32, EE), blk, 0, stream>>>(W1, w1t, DD, HH);
    transpose_f32_to_bf16<<<dim3(DD / 32, HH / 32, EE), blk, 0, stream>>>(W2, w2t, HH, DD);
    transpose_f32_to_bf16<<<dim3(GHD / 32, DD / 32, 1), blk, 0, stream>>>(Wg1, wg1t, DD, GHD);

    // 2) gating hidden: gh = relu(x @ Wg1 + bg1)   [B,GH], bf16 out
    gemm_bf16_wmma<true, true><<<dim3(GHD / BN, BQ / BM, 1), blk, 0, stream>>>(
        xbf, 0L, wg1t, 0L, bg1, 0L, ghbf, 0L, DD, GHD, DD);

    // 3) expert layer 1: h[e] = relu(x @ W1[e] + b1[e])   [E][B,H], bf16 out
    gemm_bf16_wmma<true, true><<<dim3(HH / BN, BQ / BM, EE), blk, 0, stream>>>(
        xbf, 0L, w1t, (long)HH * DD, b1, (long)HH,
        hbf, (long)BQ * HH, DD, HH, DD);

    // 4) gating tail
    gating_kernel<<<BQ / 256, blk, 0, stream>>>(ghbf, Wg2, bg2, gating, w1n, w2n);

    // 5) expert layer 2: expert_out[b][e][:] = h[e][b] @ W2[e] + b2[e]  (ldc = E*D)
    gemm_bf16_wmma<false, false><<<dim3(DD / BN, BQ / BM, EE), blk, 0, stream>>>(
        hbf, (long)BQ * HH, w2t, (long)DD * HH, b2, (long)DD,
        expert_out, (long)DD, HH, EE * DD, HH);

    // 6) gated combines
    combine_kernel<<<(BQ * (DD / 4)) / 256, blk, 0, stream>>>(
        expert_out, gating, w1n, w2n, fin, t1, t2);
}
